// IntraSpeakerTransformer_65386582114941
// MI455X (gfx1250) — matrix operand order repaired and checked by reading, not compiled, not run on hardware
//
#include <hip/hip_runtime.h>
#include <stdint.h>

#define B_  4
#define S_  2048
#define D_  512
#define H_  8
#define L_  4
#define F_  2048
#define DH_ 64
#define TD_ 1536            // 3*D
#define M_  (B_*S_)         // 8192 rows of activations
#define EPS_ 1e-5f
#define GSTRIDE 72          // 64 bf16 payload + 8 pad (144B rows, conflict-free b128 reads)
#define KSTRIDE 72          // K-tile rows: 64 feats + 8 pad
#define VSTRIDE 40          // V^T tile rows: 32 keys + 8 pad
#define PSTRIDE 40          // P tile rows: 32 keys + 8 pad

typedef __attribute__((ext_vector_type(16))) __bf16 v16bf;
typedef __attribute__((ext_vector_type(8)))  float  v8f;

union FragU {
  uint32_t u[8];
  uint4    q[2];
  v16bf    f;
};

__device__ __forceinline__ uint16_t f2bf(float x) {
  uint32_t u = __builtin_bit_cast(uint32_t, x);
  uint32_t r = (u + 0x7FFFu + ((u >> 16) & 1u)) >> 16;
  return (uint16_t)r;
}

// CDNA5 async global->LDS copy (16B per lane), tracked by ASYNCcnt.
__device__ __forceinline__ void async_ld128(uint32_t lds_off, uint64_t gaddr) {
  asm volatile("global_load_async_to_lds_b128 %0, %1, off"
               :: "v"(lds_off), "v"(gaddr)
               : "memory");
}
__device__ __forceinline__ void wait_async8() {
  asm volatile("s_wait_asynccnt 0x8" ::: "memory");
}
__device__ __forceinline__ void wait_async4() {
  asm volatile("s_wait_asynccnt 0x4" ::: "memory");
}
__device__ __forceinline__ void wait_async0() {
  asm volatile("s_wait_asynccnt 0x0" ::: "memory");
}

// ---------------------------------------------------------------- convert
__global__ void k_f32_to_bf16(const float* __restrict__ in,
                              uint16_t* __restrict__ out, long long n) {
  long long i = (long long)blockIdx.x * blockDim.x + threadIdx.x;
  long long stride = (long long)gridDim.x * blockDim.x;
  for (; i < n; i += stride) out[i] = f2bf(in[i]);
}

// ------------------------------------------------- V transpose: [B,S,3D] slice -> [B,H,DH,S]
__global__ void k_transpose_v(const uint16_t* __restrict__ qkv,
                              uint16_t* __restrict__ vt) {
  const long long n = (long long)B_ * H_ * DH_ * S_;
  long long i = (long long)blockIdx.x * blockDim.x + threadIdx.x;
  long long stride = (long long)gridDim.x * blockDim.x;
  for (; i < n; i += stride) {
    int s = (int)(i % S_);
    long long t = i / S_;
    int d = (int)(t % DH_); t /= DH_;
    int h = (int)(t % H_);
    int b = (int)(t / H_);
    vt[i] = qkv[((long long)(b * S_ + s)) * TD_ + 2 * D_ + h * DH_ + d];
  }
}

// ---------------------------------------------------------------- WMMA GEMM v3
// out[m,n] = sum_k A[m,k]*W[n,k] + bias[n]; optional ReLU; f32 and/or bf16 out.
// Block: 256 threads (8 waves). Block tile 128x128, wave tile 32x64.
// 64-deep K slabs staged through double-buffered LDS via async global->LDS.
// grid = (N/128, M/128). Kdim must be a multiple of 64.
__global__ __launch_bounds__(256) void k_gemm_bf16(
    const uint16_t* __restrict__ A, const uint16_t* __restrict__ W,
    const float* __restrict__ bias, float* __restrict__ outF,
    uint16_t* __restrict__ outB, int Kdim, int Ndim, int relu) {
  __shared__ __align__(16) uint16_t ldsA[2][128 * GSTRIDE];
  __shared__ __align__(16) uint16_t ldsB[2][128 * GSTRIDE];

  const int lane  = threadIdx.x & 31;
  const int wave  = threadIdx.x >> 5;
  const int halfl = lane >> 4;
  const int lr    = lane & 15;
  const int gm = blockIdx.y * 128;
  const int gn = blockIdx.x * 128;
  const int waveM = (wave >> 1) * 32;   // 0,32,64,96
  const int waveN = (wave & 1) * 64;    // 0,64

  // cooperative loader: 128 rows x 128B per tile = 1024 x 16B chunks, 4/thread/matrix
  auto issue = [&](int buf, int kk) {
#pragma unroll
    for (int i = 0; i < 4; ++i) {
      int c = (int)threadIdx.x + i * 256;   // 0..1023
      int row = c >> 3;                     // 0..127
      int cc  = c & 7;                      // 16B chunk within 128B row
      uint32_t la = (uint32_t)(uintptr_t)&ldsA[buf][row * GSTRIDE + cc * 8];
      uint64_t ga = (uint64_t)(uintptr_t)(A + (size_t)(gm + row) * Kdim + kk + cc * 8);
      async_ld128(la, ga);
      uint32_t lb = (uint32_t)(uintptr_t)&ldsB[buf][row * GSTRIDE + cc * 8];
      uint64_t gb = (uint64_t)(uintptr_t)(W + (size_t)(gn + row) * Kdim + kk + cc * 8);
      async_ld128(lb, gb);
    }
  };

  v8f acc[2][4] = {};
  const int nsteps = Kdim >> 6;

  issue(0, 0);
  for (int s = 0; s < nsteps; ++s) {
    const int cur = s & 1;
    if (s + 1 < nsteps) {
      issue(cur ^ 1, (s + 1) * 64);
      wait_async8();            // the 8 older ops (current buffer) are done
    } else {
      wait_async0();
    }
    __syncthreads();

    const uint16_t* Ab = ldsA[cur];
    const uint16_t* Bb = ldsB[cur];

#pragma unroll
    for (int ks = 0; ks < 64; ks += 32) {
      FragU af[2];
#pragma unroll
      for (int mi = 0; mi < 2; ++mi) {
        const uint16_t* p = Ab + (waveM + mi * 16 + lr) * GSTRIDE + ks;
        af[mi].q[0] = *(const uint4*)(p + halfl * 16);
        af[mi].q[1] = *(const uint4*)(p + halfl * 16 + 8);
      }
#pragma unroll
      for (int nj = 0; nj < 4; ++nj) {
        FragU bfr;
        const uint16_t* p = Bb + (waveN + nj * 16 + lr) * GSTRIDE + ks + halfl * 16;
        bfr.q[0] = *(const uint4*)(p);
        bfr.q[1] = *(const uint4*)(p + 8);
#pragma unroll
        for (int mi = 0; mi < 2; ++mi) {
          acc[mi][nj] = __builtin_amdgcn_wmma_f32_16x16x32_bf16(
              false, af[mi].f, false, bfr.f, (short)0, acc[mi][nj], false, false);
        }
      }
    }
    __syncthreads();
  }

#pragma unroll
  for (int mi = 0; mi < 2; ++mi)
#pragma unroll
    for (int nj = 0; nj < 4; ++nj) {
      int n = gn + waveN + nj * 16 + lr;
      float bv = bias[n];
#pragma unroll
      for (int r = 0; r < 8; ++r) {
        int m = gm + waveM + mi * 16 + halfl * 8 + r;
        float val = acc[mi][nj][r] + bv;
        if (relu) val = fmaxf(val, 0.0f);
        size_t idx = (size_t)m * Ndim + n;
        if (outF) outF[idx] = val;
        if (outB) outB[idx] = f2bf(val);
      }
    }
}

// ---------------------------------------------------------------- flash attention v2
// Block: 128 threads (4 waves). Each wave owns a 16-row q tile; block covers 64 rows.
// K and V^T tiles (32 keys) shared in double-buffered LDS via async global->LDS.
// grid = (S/64, H, B).
__global__ __launch_bounds__(128) void k_attn(
    const uint16_t* __restrict__ qkv, const uint16_t* __restrict__ vt,
    const int* __restrict__ spk, uint16_t* __restrict__ o) {
  __shared__ __align__(16) uint16_t ldsK[2][32 * KSTRIDE];   // [key][feat]
  __shared__ __align__(16) uint16_t ldsV[2][64 * VSTRIDE];   // [feat][key]
  __shared__ __align__(16) uint16_t ldsP[4][16 * PSTRIDE];   // per-wave P tile

  const int q0b = blockIdx.x * 64;
  const int h   = blockIdx.y;
  const int b   = blockIdx.z;
  const int lane  = threadIdx.x & 31;
  const int wid   = threadIdx.x >> 5;
  const int halfl = lane >> 4;
  const int lr    = lane & 15;
  const int q0w = q0b + wid * 16;

  // loader: K tile 32x128B = 256 chunks; V^T tile 64x64B = 256 chunks; 2+2 per thread
  auto issue = [&](int buf, int kb) {
#pragma unroll
    for (int i = 0; i < 2; ++i) {
      int c = (int)threadIdx.x + i * 128;       // 0..255
      int krow = c >> 3, kcc = c & 7;           // K: row 0..31, chunk 0..7
      uint32_t la = (uint32_t)(uintptr_t)&ldsK[buf][krow * KSTRIDE + kcc * 8];
      uint64_t ga = (uint64_t)(uintptr_t)(
          qkv + ((size_t)(b * S_ + kb + krow)) * TD_ + D_ + h * DH_ + kcc * 8);
      async_ld128(la, ga);
      int vrow = c >> 2, vcc = c & 3;           // V^T: row 0..63, chunk 0..3
      uint32_t lb = (uint32_t)(uintptr_t)&ldsV[buf][vrow * VSTRIDE + vcc * 8];
      uint64_t gb = (uint64_t)(uintptr_t)(
          vt + ((size_t)((b * H_ + h) * DH_ + vrow)) * S_ + kb + vcc * 8);
      async_ld128(lb, gb);
    }
  };

  // Q A-fragments (16x64 split into two 16x32), straight from global (read once)
  const uint32_t* qrow =
      (const uint32_t*)(qkv + ((size_t)(b * S_ + q0w + lr)) * TD_ + h * DH_);
  FragU qa[2];
#pragma unroll
  for (int t = 0; t < 2; ++t)
#pragma unroll
    for (int v = 0; v < 8; ++v) {
      int k = t * 32 + halfl * 16 + v * 2;
      qa[t].u[v] = qrow[k >> 1];
    }

  int sq[8];
#pragma unroll
  for (int r = 0; r < 8; ++r) sq[r] = spk[b * S_ + q0w + halfl * 8 + r];

  v8f oacc[4] = {};
  float mrow[8], lrow[8];
#pragma unroll
  for (int r = 0; r < 8; ++r) { mrow[r] = -3e38f; lrow[r] = 0.0f; }

  const int nchunks = (q0b + 64 + 31) >> 5;
  issue(0, 0);
  for (int kt = 0; kt < nchunks; ++kt) {
    const int kb = kt * 32;
    const int cur = kt & 1;
    if (kt + 1 < nchunks) {
      issue(cur ^ 1, (kt + 1) * 32);
      wait_async4();
    } else {
      wait_async0();
    }
    __syncthreads();

    if (kb <= q0w + 15) {      // chunk overlaps this wave's causal range
      const uint16_t* Kb = ldsK[cur];
      const uint16_t* Vb = ldsV[cur];
      uint16_t* Pw = ldsP[wid];

      // scores for two 16-key subtiles from shared K tile
      v8f sc[2] = {};
#pragma unroll
      for (int sub = 0; sub < 2; ++sub) {
#pragma unroll
        for (int t = 0; t < 2; ++t) {
          FragU kf;
          const uint16_t* p = Kb + (sub * 16 + lr) * KSTRIDE + t * 32 + halfl * 16;
          kf.q[0] = *(const uint4*)(p);
          kf.q[1] = *(const uint4*)(p + 8);
          sc[sub] = __builtin_amdgcn_wmma_f32_16x16x32_bf16(
              false, qa[t].f, false, kf.f, (short)0, sc[sub], false, false);
        }
      }

      const int sk0 = spk[b * S_ + kb + lr];
      const int sk1 = spk[b * S_ + kb + 16 + lr];

      float p0[8], p1[8], corr[8];
#pragma unroll
      for (int r = 0; r < 8; ++r) {
        const int qrow_i = q0w + halfl * 8 + r;
        float s0 = sc[0][r] * 0.125f;
        float s1 = sc[1][r] * 0.125f;
        const bool a0 = (kb + lr <= qrow_i) && (sk0 == sq[r]);
        const bool a1 = (kb + 16 + lr <= qrow_i) && (sk1 == sq[r]);
        if (!a0) s0 = -3e38f;
        if (!a1) s1 = -3e38f;
        float mx = fmaxf(s0, s1);
#pragma unroll
        for (int off = 1; off < 16; off <<= 1)
          mx = fmaxf(mx, __shfl_xor(mx, off, 16));
        const float mnew = fmaxf(mrow[r], mx);
        p0[r] = a0 ? __expf(s0 - mnew) : 0.0f;
        p1[r] = a1 ? __expf(s1 - mnew) : 0.0f;
        float ssum = p0[r] + p1[r];
#pragma unroll
        for (int off = 1; off < 16; off <<= 1)
          ssum += __shfl_xor(ssum, off, 16);
        const float c = __expf(mrow[r] - mnew);
        lrow[r] = lrow[r] * c + ssum;
        mrow[r] = mnew;
        corr[r] = c;
      }
#pragma unroll
      for (int j = 0; j < 4; ++j)
#pragma unroll
        for (int r = 0; r < 8; ++r) oacc[j][r] *= corr[r];

      // P (16x32) -> per-wave LDS tile, reload as A-fragment (in-wave, no barrier)
#pragma unroll
      for (int r = 0; r < 8; ++r) {
        const int m = halfl * 8 + r;
        Pw[m * PSTRIDE + lr]      = f2bf(p0[r]);
        Pw[m * PSTRIDE + 16 + lr] = f2bf(p1[r]);
      }
      FragU pa;
      {
        const uint16_t* p = Pw + lr * PSTRIDE;
        pa.q[0] = *(const uint4*)(p + halfl * 16);
        pa.q[1] = *(const uint4*)(p + halfl * 16 + 8);
      }
      // P @ V from shared V^T tile
#pragma unroll
      for (int j = 0; j < 4; ++j) {
        FragU vb;
        const uint16_t* p = Vb + (j * 16 + lr) * VSTRIDE + halfl * 16;
        vb.q[0] = *(const uint4*)(p);
        vb.q[1] = *(const uint4*)(p + 8);
        oacc[j] = __builtin_amdgcn_wmma_f32_16x16x32_bf16(
            false, pa.f, false, vb.f, (short)0, oacc[j], false, false);
      }
    }
    __syncthreads();
  }

  float inv[8];
#pragma unroll
  for (int r = 0; r < 8; ++r) inv[r] = 1.0f / lrow[r];
#pragma unroll
  for (int j = 0; j < 4; ++j)
#pragma unroll
    for (int r = 0; r < 8; ++r) {
      const int row = q0w + halfl * 8 + r;
      const int col = h * DH_ + j * 16 + lr;
      o[((size_t)(b * S_ + row)) * D_ + col] = f2bf(oacc[j][r] * inv[r]);
    }
}

// ---------------------------------------------------------------- residual + LayerNorm
__global__ __launch_bounds__(256) void k_resid_ln(
    const float* __restrict__ x, const float* __restrict__ a,
    const float* __restrict__ g, const float* __restrict__ bb,
    float* __restrict__ xout, uint16_t* __restrict__ bout) {
  __shared__ float red[256];
  const int row = blockIdx.x;
  const int t = threadIdx.x;
  const float* xr = x + (size_t)row * D_;
  const float* ar = a + (size_t)row * D_;
  float v0 = xr[t] + ar[t];
  float v1 = xr[t + 256] + ar[t + 256];
  red[t] = v0 + v1;
  __syncthreads();
  for (int o = 128; o > 0; o >>= 1) {
    if (t < o) red[t] += red[t + o];
    __syncthreads();
  }
  const float mu = red[0] * (1.0f / D_);
  __syncthreads();
  const float d0 = v0 - mu, d1 = v1 - mu;
  red[t] = d0 * d0 + d1 * d1;
  __syncthreads();
  for (int o = 128; o > 0; o >>= 1) {
    if (t < o) red[t] += red[t + o];
    __syncthreads();
  }
  const float rs = rsqrtf(red[0] * (1.0f / D_) + EPS_);
  const float y0 = d0 * rs * g[t] + bb[t];
  const float y1 = d1 * rs * g[t + 256] + bb[t + 256];
  xout[(size_t)row * D_ + t] = y0;
  xout[(size_t)row * D_ + t + 256] = y1;
  bout[(size_t)row * D_ + t] = f2bf(y0);
  bout[(size_t)row * D_ + t + 256] = f2bf(y1);
}

// ---------------------------------------------------------------- host
extern "C" void kernel_launch(void* const* d_in, const int* in_sizes, int n_in,
                              void* d_out, int out_size, void* d_ws, size_t ws_size,
                              hipStream_t stream) {
  (void)in_sizes; (void)n_in; (void)out_size; (void)ws_size;
  const float* emb  = (const float*)d_in[0];
  const float* Wqkv = (const float*)d_in[1];
  const float* bqkv = (const float*)d_in[2];
  const float* Wo   = (const float*)d_in[3];
  const float* bo   = (const float*)d_in[4];
  const float* W1   = (const float*)d_in[5];
  const float* b1   = (const float*)d_in[6];
  const float* W2   = (const float*)d_in[7];
  const float* b2   = (const float*)d_in[8];
  const float* ln1g = (const float*)d_in[9];
  const float* ln1b = (const float*)d_in[10];
  const float* ln2g = (const float*)d_in[11];
  const float* ln2b = (const float*)d_in[12];
  const int*   spk  = (const int*)d_in[13];

  char* ws = (char*)d_ws;
  size_t off = 0;
  auto alloc = [&](size_t bytes) -> char* {
    char* p = ws + off;
    off += (bytes + 255) & ~(size_t)255;
    return p;
  };
  float*    x     = (float*)   alloc((size_t)M_ * D_ * 4);
  float*    aout  = (float*)   alloc((size_t)M_ * D_ * 4);
  uint16_t* xb    = (uint16_t*)alloc((size_t)M_ * D_ * 2);
  uint16_t* qkvb  = (uint16_t*)alloc((size_t)M_ * TD_ * 2);
  uint16_t* vt    = (uint16_t*)alloc((size_t)B_ * H_ * DH_ * S_ * 2);
  uint16_t* ob    = (uint16_t*)alloc((size_t)M_ * D_ * 2);
  uint16_t* hb    = (uint16_t*)alloc((size_t)M_ * F_ * 2);
  uint16_t* wqkvb = (uint16_t*)alloc((size_t)L_ * TD_ * D_ * 2);
  uint16_t* wob   = (uint16_t*)alloc((size_t)L_ * D_ * D_ * 2);
  uint16_t* w1b   = (uint16_t*)alloc((size_t)L_ * F_ * D_ * 2);
  uint16_t* w2b   = (uint16_t*)alloc((size_t)L_ * D_ * F_ * 2);

  // weights -> bf16 (every call; deterministic)
  k_f32_to_bf16<<<2048, 256, 0, stream>>>(Wqkv, wqkvb, (long long)L_ * TD_ * D_);
  k_f32_to_bf16<<<2048, 256, 0, stream>>>(Wo,   wob,   (long long)L_ * D_ * D_);
  k_f32_to_bf16<<<2048, 256, 0, stream>>>(W1,   w1b,   (long long)L_ * F_ * D_);
  k_f32_to_bf16<<<2048, 256, 0, stream>>>(W2,   w2b,   (long long)L_ * D_ * F_);

  // x = embeddings
  hipMemcpyAsync(x, emb, (size_t)M_ * D_ * 4, hipMemcpyDeviceToDevice, stream);
  k_f32_to_bf16<<<2048, 256, 0, stream>>>(x, xb, (long long)M_ * D_);

  for (int l = 0; l < L_; ++l) {
    // QKV projection (bf16 out only)
    k_gemm_bf16<<<dim3(TD_ / 128, M_ / 128), 256, 0, stream>>>(
        xb, wqkvb + (size_t)l * TD_ * D_, bqkv + (size_t)l * TD_,
        nullptr, qkvb, D_, TD_, 0);
    // V transpose for attention B-fragments
    k_transpose_v<<<2048, 256, 0, stream>>>(qkvb, vt);
    // masked flash attention (shared K/V^T tiles, async double buffering)
    k_attn<<<dim3(S_ / 64, H_, B_), 128, 0, stream>>>(qkvb, vt, spk, ob);
    // output projection (f32 out for residual)
    k_gemm_bf16<<<dim3(D_ / 128, M_ / 128), 256, 0, stream>>>(
        ob, wob + (size_t)l * D_ * D_, bo + (size_t)l * D_,
        aout, nullptr, D_, D_, 0);
    // x = LN1(x + attn)
    k_resid_ln<<<M_, 256, 0, stream>>>(x, aout, ln1g + (size_t)l * D_,
                                       ln1b + (size_t)l * D_, x, xb);
    // FFN up + ReLU (bf16 out)
    k_gemm_bf16<<<dim3(F_ / 128, M_ / 128), 256, 0, stream>>>(
        xb, w1b + (size_t)l * F_ * D_, b1 + (size_t)l * F_,
        nullptr, hb, D_, F_, 1);
    // FFN down (f32 out)
    k_gemm_bf16<<<dim3(D_ / 128, M_ / 128), 256, 0, stream>>>(
        hb, w2b + (size_t)l * D_ * F_, b2 + (size_t)l * D_,
        aout, nullptr, F_, D_, 0);
    // x = LN2(x + ffn)
    k_resid_ln<<<M_, 256, 0, stream>>>(x, aout, ln2g + (size_t)l * D_,
                                       ln2b + (size_t)l * D_, x, xb);
  }

  hipMemcpyAsync(d_out, x, (size_t)M_ * D_ * 4, hipMemcpyDeviceToDevice, stream);
}
